// fullEleAtt_32993938768180
// MI455X (gfx1250) — compile-verified, matrix operand order checked by reading, not run on hardware
//
#include <hip/hip_runtime.h>

#define CC 256
#define HH 96
#define WW 96
#define BB 16
#define SP (HH * WW)                       // 9216 spatial positions
#define XSZ ((size_t)BB * CC * SP)         // 37,748,736 elements

typedef __attribute__((ext_vector_type(16))) __bf16 v16bf;
typedef __attribute__((ext_vector_type(8)))  float  v8f;

union Frag {
    v16bf v;
    uint4 u[2];
};

__device__ __forceinline__ unsigned short f2bf(float f) {
    unsigned u = __float_as_uint(f);
    u += 0x7FFFu + ((u >> 16) & 1u);       // round-to-nearest-even
    return (unsigned short)(u >> 16);
}

// Per-lane bf16 fragment: two runs of 8 consecutive K (at +0 and +16 elements).
__device__ __forceinline__ Frag load_frag(const unsigned short* p) {
    Frag f;
    f.u[0] = *(const uint4*)(p);
    f.u[1] = *(const uint4*)(p + 16);
    return f;
}

__device__ __forceinline__ v8f wmma_bf16(const Frag& a, const Frag& b, v8f c) {
    return __builtin_amdgcn_wmma_f32_16x16x32_bf16(
        /*neg_a=*/false, a.v, /*neg_b=*/false, b.v,
        /*c_mod=*/(short)0, c, /*reuse_a=*/false, /*reuse_b=*/false);
}

// ---------------------------------------------------------------------------
// Weights fp32 -> bf16 (theta | phi | g concatenated)
// ---------------------------------------------------------------------------
__global__ void k_cvt_w(const float* __restrict__ wt, const float* __restrict__ wp,
                        const float* __restrict__ wg, unsigned short* __restrict__ wbf) {
    int t = blockIdx.x * 256 + threadIdx.x;        // 0 .. 196607
    if (t < 65536)        wbf[t] = f2bf(wt[t]);
    else if (t < 131072)  wbf[t] = f2bf(wp[t - 65536]);
    else                  wbf[t] = f2bf(wg[t - 131072]);
}

// ---------------------------------------------------------------------------
// x [b,c,s] fp32 -> XT [b,s,c] bf16 (LDS tile transpose, coalesced both sides)
// ---------------------------------------------------------------------------
__global__ void k_transpose(const float* __restrict__ x, unsigned short* __restrict__ xt) {
    __shared__ float tile[32][33];
    int tx = threadIdx.x, ty = threadIdx.y;        // 32 x 8
    int s0 = blockIdx.x * 32, c0 = blockIdx.y * 32, b = blockIdx.z;
    const float* xb = x + (size_t)b * CC * SP;
    unsigned short* xtb = xt + (size_t)b * SP * CC;
#pragma unroll
    for (int r = 0; r < 4; ++r)
        tile[ty + r * 8][tx] = xb[(size_t)(c0 + ty + r * 8) * SP + s0 + tx];
    __syncthreads();
#pragma unroll
    for (int r = 0; r < 4; ++r)
        xtb[(size_t)(s0 + ty + r * 8) * CC + c0 + tx] = f2bf(tile[tx][ty + r * 8]);
}

// ---------------------------------------------------------------------------
// Fused 1x1-conv theta/phi/g:  Y[s,o] = sum_c XT[b,s,c] * W[o,c] + bias[o]
// A = XT tile rows (M = s, k contiguous), B = weight rows (N = o, k contiguous).
// Two s-tiles per iteration share the 3 weight fragments: 5 loads / 6 WMMA.
// Bias is lane-invariant (o = o0 + lane%16) and hoisted.
// Results scattered to Q/K [b,p,f] and Vt [b,f,p] as bf16.
// ---------------------------------------------------------------------------
template <int ROUND>
__global__ void __launch_bounds__(128, 2)
k_conv_qkv(const unsigned short* __restrict__ XT,
           const unsigned short* __restrict__ wbf,
           const float* __restrict__ bt, const float* __restrict__ bp,
           const float* __restrict__ bg,
           unsigned short* __restrict__ Qb,
           unsigned short* __restrict__ Kb,
           unsigned short* __restrict__ Vt) {
    constexpr int N_ = (ROUND == 2) ? CC : WW;       // attention axis length
    constexpr int F_ = (ROUND == 2) ? SP : CC * HH;  // feature (contraction) length
    int wave = threadIdx.x >> 5, lane = threadIdx.x & 31;
    int m = lane & 15, g = lane >> 4;
    int o0 = blockIdx.y * 16;
    int b  = blockIdx.z;
    int wid = blockIdx.x * 4 + wave;                 // 0..47 wave slots
    int o = o0 + m;                                  // C-frag: N index = o (lane-fixed)
    float biasT = bt[o], biasP = bp[o], biasG = bg[o];
    size_t wrow = (size_t)o * CC;
    const unsigned short* wt  = wbf;
    const unsigned short* wp  = wbf + 65536;
    const unsigned short* wgm = wbf + 131072;
    const unsigned short* xtb = XT + (size_t)b * SP * CC;

    for (int tp = 0; tp < 6; ++tp) {                 // 48 waves * 6 pairs = 288 pairs
        int sA = (wid * 6 + tp) * 32;                // two 16-wide s-tiles: sA, sA+16
        v8f ct0 = {}, cp0 = {}, cg0 = {};
        v8f ct1 = {}, cp1 = {}, cg1 = {};
#pragma unroll
        for (int ki = 0; ki < 8; ++ki) {             // K = 256 = 8 * 32
            int kb = ki * 32 + g * 8;
            Frag a0  = load_frag(xtb + (size_t)(sA + m) * CC + kb);
            Frag a1  = load_frag(xtb + (size_t)(sA + 16 + m) * CC + kb);
            Frag bT  = load_frag(wt  + wrow + kb);
            Frag bP  = load_frag(wp  + wrow + kb);
            Frag bG  = load_frag(wgm + wrow + kb);
            ct0 = wmma_bf16(a0, bT, ct0);
            cp0 = wmma_bf16(a0, bP, cp0);
            cg0 = wmma_bf16(a0, bG, cg0);
            ct1 = wmma_bf16(a1, bT, ct1);
            cp1 = wmma_bf16(a1, bP, cp1);
            cg1 = wmma_bf16(a1, bG, cg1);
        }
#pragma unroll
        for (int half = 0; half < 2; ++half) {
            const v8f& ct = half ? ct1 : ct0;
            const v8f& cp = half ? cp1 : cp0;
            const v8f& cg = half ? cg1 : cg0;
#pragma unroll
            for (int r = 0; r < 8; ++r) {
                int s = sA + half * 16 + r + 8 * g;  // C-frag: M index = s
                int p, f;
                if (ROUND == 0)      { p = s % WW; f = o * HH + s / WW; }   // width
                else if (ROUND == 1) { p = s / WW; f = o * WW + s % WW; }   // height
                else                 { p = o;      f = s; }                 // channel
                size_t qi = ((size_t)b * N_ + p) * F_ + f;
                size_t vi = ((size_t)b * F_ + f) * (size_t)N_ + p;
                Qb[qi] = f2bf(ct[r] + biasT);
                Kb[qi] = f2bf(cp[r] + biasP);
                Vt[vi] = f2bf(cg[r] + biasG);
            }
        }
    }
}

// ---------------------------------------------------------------------------
// Scores S[b,i,j] = sum_f Q[b,i,f]*K[b,j,f].  32x32 tile per block; 4 waves
// split the (huge) K dim, partials reduced through LDS.
// ---------------------------------------------------------------------------
template <int N_, int F_>
__global__ void __launch_bounds__(128, 2)
k_score(const unsigned short* __restrict__ Q,
        const unsigned short* __restrict__ Kb,
        float* __restrict__ S_) {
    __shared__ float red[4][4][256];                 // 16 KB
    int wave = threadIdx.x >> 5, lane = threadIdx.x & 31;
    int m = lane & 15, g = lane >> 4;
    int i0 = blockIdx.y * 32, j0 = blockIdx.x * 32, b = blockIdx.z;
    constexpr int KS = F_ / 4;
    int k0 = wave * KS;
    v8f c00 = {}, c01 = {}, c10 = {}, c11 = {};
    const unsigned short* qb = Q  + (size_t)b * N_ * F_;
    const unsigned short* kB = Kb + (size_t)b * N_ * F_;
#pragma unroll 2
    for (int k = k0; k < k0 + KS; k += 32) {
        int kk = k + g * 8;
        Frag a0 = load_frag(qb + (size_t)(i0 + m) * F_ + kk);
        Frag a1 = load_frag(qb + (size_t)(i0 + 16 + m) * F_ + kk);
        Frag b0 = load_frag(kB + (size_t)(j0 + m) * F_ + kk);
        Frag b1 = load_frag(kB + (size_t)(j0 + 16 + m) * F_ + kk);
        c00 = wmma_bf16(a0, b0, c00);
        c01 = wmma_bf16(a0, b1, c01);
        c10 = wmma_bf16(a1, b0, c10);
        c11 = wmma_bf16(a1, b1, c11);
    }
#pragma unroll
    for (int r = 0; r < 8; ++r) {
        red[wave][0][r * 32 + lane] = c00[r];
        red[wave][1][r * 32 + lane] = c01[r];
        red[wave][2][r * 32 + lane] = c10[r];
        red[wave][3][r * 32 + lane] = c11[r];
    }
    __syncthreads();
    for (int e = threadIdx.x; e < 1024; e += 128) {
        int fi = e >> 8, rem = e & 255, r = rem >> 5, l = rem & 31;
        float v = red[0][fi][rem] + red[1][fi][rem] + red[2][fi][rem] + red[3][fi][rem];
        int mi = i0 + ((fi >> 1) << 4) + r + 8 * (l >> 4);
        int nj = j0 + ((fi & 1) << 4) + (l & 15);
        S_[((size_t)b * N_ + mi) * N_ + nj] = v;
    }
}

// ---------------------------------------------------------------------------
// Row softmax (fp32) -> P bf16.  One 128-thread block per (b, row); n <= 256.
// ---------------------------------------------------------------------------
__global__ void k_softmax(const float* __restrict__ S_, unsigned short* __restrict__ P, int n_) {
    __shared__ float red[128];
    int t = threadIdx.x;
    int i = blockIdx.x, b = blockIdx.y;
    const float* row = S_ + ((size_t)b * n_ + i) * n_;
    float v0 = (t < n_)       ? row[t]       : -3.4e38f;
    float v1 = (t + 128 < n_) ? row[t + 128] : -3.4e38f;
    red[t] = fmaxf(v0, v1);
    __syncthreads();
    for (int s = 64; s > 0; s >>= 1) {
        if (t < s) red[t] = fmaxf(red[t], red[t + s]);
        __syncthreads();
    }
    float mx = red[0];
    __syncthreads();
    float e0 = (t < n_)       ? __expf(v0 - mx) : 0.f;
    float e1 = (t + 128 < n_) ? __expf(v1 - mx) : 0.f;
    red[t] = e0 + e1;
    __syncthreads();
    for (int s = 64; s > 0; s >>= 1) {
        if (t < s) red[t] += red[t + s];
        __syncthreads();
    }
    float inv = 1.0f / red[0];
    unsigned short* prow = P + ((size_t)b * n_ + i) * n_;
    if (t < n_)       prow[t]       = f2bf(e0 * inv);
    if (t + 128 < n_) prow[t + 128] = f2bf(e1 * inv);
}

// ---------------------------------------------------------------------------
// out[b,i,f] = sum_j P[b,i,j] * Vt[b,f,j]; fused residual:
// x_next[idx(p,f)] = x_prev[idx] + beta * out.  (p,f)->(c,h,w) is a bijection,
// so each element is touched exactly once (in-place safe).
// A fragments (P rows) hoisted; two f-tiles per iteration for ILP.
// ---------------------------------------------------------------------------
template <int ROUND>
__global__ void __launch_bounds__(128, 2)
k_out(const unsigned short* __restrict__ P,
      const unsigned short* __restrict__ Vt,
      const float* __restrict__ xprev, float* __restrict__ xnext,
      const float* __restrict__ beta_p) {
    constexpr int N_ = (ROUND == 2) ? CC : WW;
    constexpr int F_ = (ROUND == 2) ? SP : CC * HH;
    constexpr int KI = N_ / 32;
    int wave = threadIdx.x >> 5, lane = threadIdx.x & 31;
    int m = lane & 15, g = lane >> 4;
    int b  = blockIdx.z;
    int i0 = blockIdx.y * 16;
    int wid = blockIdx.x * 4 + wave;                 // 0..47
    float beta = *beta_p;

    const unsigned short* pr = P + ((size_t)b * N_ + i0 + m) * N_ + g * 8;
    Frag a[KI];
#pragma unroll
    for (int ki = 0; ki < KI; ++ki)
        a[ki] = load_frag(pr + ki * 32);

    const unsigned short* vb = Vt + (size_t)b * F_ * (size_t)N_;
    for (int fp = wid; fp < F_ / 32; fp += 48) {     // pairs of 16-wide f-tiles
        int f0 = fp * 32 + m;
        int f1 = f0 + 16;
        const unsigned short* vp0 = vb + (size_t)f0 * N_ + g * 8;
        const unsigned short* vp1 = vb + (size_t)f1 * N_ + g * 8;
        v8f c0 = {}, c1 = {};
#pragma unroll
        for (int ki = 0; ki < KI; ++ki) {
            Frag b0 = load_frag(vp0 + ki * 32);
            Frag b1 = load_frag(vp1 + ki * 32);
            c0 = wmma_bf16(a[ki], b0, c0);
            c1 = wmma_bf16(a[ki], b1, c1);
        }
#pragma unroll
        for (int half = 0; half < 2; ++half) {
            int f = half ? f1 : f0;                  // C-frag: N index = f
            const v8f& c = half ? c1 : c0;
#pragma unroll
            for (int r = 0; r < 8; ++r) {
                int p = i0 + r + 8 * g;              // C-frag: M index = axis pos
                int cc, hh, ww;
                if (ROUND == 0)      { ww = p; cc = f / HH; hh = f % HH; }
                else if (ROUND == 1) { hh = p; cc = f / WW; ww = f % WW; }
                else                 { cc = p; hh = f / WW; ww = f % WW; }
                size_t idx = (((size_t)b * CC + cc) * HH + hh) * WW + ww;
                xnext[idx] = xprev[idx] + beta * c[r];
            }
        }
    }
}

// ---------------------------------------------------------------------------
extern "C" void kernel_launch(void* const* d_in, const int* in_sizes, int n_in,
                              void* d_out, int out_size, void* d_ws, size_t ws_size,
                              hipStream_t stream) {
    (void)in_sizes; (void)n_in; (void)out_size; (void)ws_size;
    const float* x_in = (const float*)d_in[0];
    const float* w_t  = (const float*)d_in[1];
    const float* b_t  = (const float*)d_in[2];
    const float* w_p  = (const float*)d_in[3];
    const float* b_p  = (const float*)d_in[4];
    const float* w_g  = (const float*)d_in[5];
    const float* b_g  = (const float*)d_in[6];
    const float* betas[3] = { (const float*)d_in[7], (const float*)d_in[8], (const float*)d_in[9] };

    char* wsp = (char*)d_ws;
    auto take = [&](size_t bytes) { char* r = wsp; wsp += (bytes + 255) & ~(size_t)255; return r; };
    float*          x1  = (float*)take(XSZ * 4);
    unsigned short* XT  = (unsigned short*)take(XSZ * 2);
    unsigned short* Qb  = (unsigned short*)take(XSZ * 2);
    unsigned short* Kb  = (unsigned short*)take(XSZ * 2);
    unsigned short* Vt  = (unsigned short*)take(XSZ * 2);
    float*          Sb  = (float*)take((size_t)BB * CC * CC * 4);
    unsigned short* Pb  = (unsigned short*)take((size_t)BB * CC * CC * 2);
    unsigned short* wbf = (unsigned short*)take((size_t)3 * 65536 * 2);

    k_cvt_w<<<dim3(768), dim3(256), 0, stream>>>(w_t, w_p, w_g, wbf);

    for (int r = 0; r < 3; ++r) {
        const float* xsrc = (r == 0) ? x_in : x1;
        float*       xdst = (r == 2) ? (float*)d_out : x1;
        int n_ = (r == 2) ? CC : WW;

        k_transpose<<<dim3(SP / 32, CC / 32, BB), dim3(32, 8), 0, stream>>>(xsrc, XT);

        if (r == 0)
            k_conv_qkv<0><<<dim3(12, 16, BB), dim3(128), 0, stream>>>(XT, wbf, b_t, b_p, b_g, Qb, Kb, Vt);
        else if (r == 1)
            k_conv_qkv<1><<<dim3(12, 16, BB), dim3(128), 0, stream>>>(XT, wbf, b_t, b_p, b_g, Qb, Kb, Vt);
        else
            k_conv_qkv<2><<<dim3(12, 16, BB), dim3(128), 0, stream>>>(XT, wbf, b_t, b_p, b_g, Qb, Kb, Vt);

        if (r == 2)
            k_score<256, SP><<<dim3(8, 8, BB), dim3(128), 0, stream>>>(Qb, Kb, Sb);
        else
            k_score<96, CC * HH><<<dim3(3, 3, BB), dim3(128), 0, stream>>>(Qb, Kb, Sb);

        k_softmax<<<dim3(n_, BB), dim3(128), 0, stream>>>(Sb, Pb, n_);

        if (r == 0)
            k_out<0><<<dim3(12, 6, BB), dim3(128), 0, stream>>>(Pb, Vt, xsrc, xdst, betas[0]);
        else if (r == 1)
            k_out<1><<<dim3(12, 6, BB), dim3(128), 0, stream>>>(Pb, Vt, xsrc, xdst, betas[1]);
        else
            k_out<2><<<dim3(12, 16, BB), dim3(128), 0, stream>>>(Pb, Vt, xsrc, xdst, betas[2]);
    }
}
